// OwnGATConv_57561151701650
// MI455X (gfx1250) — matrix operand order repaired and checked by reading, not compiled, not run on hardware
//
#include <hip/hip_runtime.h>

// ---------------------------------------------------------------------------
// GAT layer, fused, for MI455X (gfx1250, wave32, WMMA, async-to-LDS)
// N=8192, IN_F=OUT_F=256, ALPHA=0.2
//
// Single-pass softmax: shift with conservative per-row upper bound
//   M_i = max(0, leaky_relu(s1[i] + max_j s2[j]))
// (valid since leaky_relu is monotone; shift cancels in softmax). The
// denominator is accumulated from the same coefficients that feed the WMMA
// A-fragments, so adj (268 MB) is streamed exactly once: ~11.5us at 23.3TB/s.
// ---------------------------------------------------------------------------

typedef __attribute__((ext_vector_type(16))) __bf16 v16bf;
typedef __attribute__((ext_vector_type(8)))  float  v8f;

union BF16x16 {
    uint4        q[2];
    unsigned     u[8];
    v16bf        v;
};

__device__ __forceinline__ unsigned short f2bf(float x) {
    unsigned u = __builtin_bit_cast(unsigned, x);
    unsigned r = (u + 0x7fffu + ((u >> 16) & 1u)) >> 16;  // round-to-nearest-even
    return (unsigned short)r;
}

#define NROW 8192
#define NF   256
#define ALPHA 0.2f

// ---------------------------------------------------------------------------
// Kernel 0: convert x (8192x256) and w (256x256) f32 -> bf16
// ---------------------------------------------------------------------------
__global__ void gat_convert(const float* __restrict__ x, const float* __restrict__ w,
                            unsigned short* __restrict__ xb, unsigned short* __restrict__ wb,
                            int nx, int nw) {
    int i = blockIdx.x * 256 + threadIdx.x;
    if (i < nx)            xb[i] = f2bf(x[i]);
    else if (i - nx < nw)  wb[i - nx] = f2bf(w[i - nx]);
}

// ---------------------------------------------------------------------------
// Kernel 1: h = x @ w^T via v_wmma_f32_16x16x32_bf16.
// One wave per 16x16 output tile; K = 256 in 8 chunks of 32.
// Writes h (f32, row-major) and ht (bf16, transposed: ht[f*8192 + i] = h[i,f]).
// ---------------------------------------------------------------------------
__global__ void gat_h_gemm(const unsigned short* __restrict__ xb,
                           const unsigned short* __restrict__ wb,
                           float* __restrict__ h, unsigned short* __restrict__ ht) {
    const int lane = threadIdx.x & 31, wave = threadIdx.x >> 5;
    const int r15 = lane & 15, hi = lane >> 4, g = hi * 8;
    const int tile = blockIdx.x * 8 + wave;          // 8192 tiles total
    const int rowBase = (tile >> 4) * 16;            // 512 row tiles
    const int colBase = (tile & 15) * 16;            // 16 col tiles

    const unsigned short* arow = xb + (rowBase + r15) * NF;
    const unsigned short* brow = wb + (colBase + r15) * NF;

    v8f acc = {0.f, 0.f, 0.f, 0.f, 0.f, 0.f, 0.f, 0.f};

#pragma unroll
    for (int kc = 0; kc < 8; ++kc) {
        const int k0 = kc * 32;
        BF16x16 A, B;
        // A (16x32 bf16): lane<16 -> K = {0..7,16..23}; lane>=16 -> K = {8..15,24..31}
        A.q[0] = *(const uint4*)(arow + k0 + g);
        A.q[1] = *(const uint4*)(arow + k0 + 16 + g);
        // B (32x16 bf16): lane<16 -> K = {0..7,16..23}; lane>=16 -> K = {8..15,24..31}
        B.q[0] = *(const uint4*)(brow + k0 + g);
        B.q[1] = *(const uint4*)(brow + k0 + 16 + g);
        acc = __builtin_amdgcn_wmma_f32_16x16x32_bf16(false, A.v, false, B.v,
                                                      (short)0, acc, false, false);
    }

    // D layout: VGPR r -> M = r + 8*hi, N = lane&15
#pragma unroll
    for (int r = 0; r < 8; ++r)
        h[(rowBase + 8 * hi + r) * NF + colBase + r15] = acc[r];

    // transposed bf16 copy: 8 consecutive rows, same column -> one b128 store
    uint4 p;
    unsigned* pu = (unsigned*)&p;
#pragma unroll
    for (int q = 0; q < 4; ++q)
        pu[q] = (unsigned)f2bf(acc[2 * q]) | ((unsigned)f2bf(acc[2 * q + 1]) << 16);
    *(uint4*)(ht + (colBase + r15) * NROW + rowBase + 8 * hi) = p;
}

// ---------------------------------------------------------------------------
// Kernel 2: s1[i] = h[i,:]@a1, s2[i] = h[i,:]@a2. One wave per row.
// ---------------------------------------------------------------------------
__global__ void gat_scores(const float* __restrict__ h, const float* __restrict__ a,
                           float* __restrict__ s1, float* __restrict__ s2) {
    const int lane = threadIdx.x & 31, wave = threadIdx.x >> 5;
    const int row = blockIdx.x * 8 + wave;
    const float* hr = h + row * NF;
    float acc1 = 0.f, acc2 = 0.f;
#pragma unroll
    for (int q = 0; q < 8; ++q) {
        const int f = lane + 32 * q;
        const float hv = hr[f];
        acc1 += hv * a[f];
        acc2 += hv * a[NF + f];
    }
#pragma unroll
    for (int off = 16; off; off >>= 1) {
        acc1 += __shfl_xor(acc1, off);
        acc2 += __shfl_xor(acc2, off);
    }
    if (lane == 0) { s1[row] = acc1; s2[row] = acc2; }
}

// ---------------------------------------------------------------------------
// Kernel 2b: S2MAX = max_j s2[j] (single block)
// ---------------------------------------------------------------------------
__global__ void gat_s2max(const float* __restrict__ s2, float* __restrict__ s2max) {
    __shared__ float red[256];
    const int tid = threadIdx.x;
    float m = -1e30f;
    for (int t = tid; t < NROW; t += 256) m = fmaxf(m, s2[t]);
    red[tid] = m; __syncthreads();
    for (int s = 128; s; s >>= 1) {
        if (tid < s) red[tid] = fmaxf(red[tid], red[tid + s]);
        __syncthreads();
    }
    if (tid == 0) *s2max = red[0];
}

// ---------------------------------------------------------------------------
// Async staging: ht[:, j0:j0+32] -> LDS buffer (16 KB), 128 threads.
// global_load_async_to_lds_b128 applies INST_OFFSET to both global and LDS
// addresses (ISA 10.7 / 15.18), so one base pair + 4 offsets per feature row.
// Tracked by ASYNCcnt.
// ---------------------------------------------------------------------------
__device__ __forceinline__ void issue_stage(const unsigned short* __restrict__ ht,
                                            int j0, unsigned short* dst, int tid) {
#pragma unroll
    for (int ff = 0; ff < 2; ++ff) {
        const int f = tid + ff * 128;
        const unsigned long long gp = (unsigned long long)(size_t)(ht + f * NROW + j0);
        const unsigned lds = (unsigned)(size_t)(dst + f * 32);   // low 32 bits = LDS offset
        asm volatile("global_load_async_to_lds_b128 %0, %1, off"           :: "v"(lds), "v"(gp) : "memory");
        asm volatile("global_load_async_to_lds_b128 %0, %1, off offset:16" :: "v"(lds), "v"(gp) : "memory");
        asm volatile("global_load_async_to_lds_b128 %0, %1, off offset:32" :: "v"(lds), "v"(gp) : "memory");
        asm volatile("global_load_async_to_lds_b128 %0, %1, off offset:48" :: "v"(lds), "v"(gp) : "memory");
    }
}

// ---------------------------------------------------------------------------
// Kernel 3: fused attention GEMM + single-pass softmax + elu.
// Block = 128 threads (4 waves); each wave owns 16 rows x all 256 cols.
// Per 32-column K-chunk: attention coefficients built in registers (exp with
// conservative shift), packed into the ISA A-fragment layout, row-denominator
// accumulated alongside; B fragments double-buffered in LDS via async copies.
// ---------------------------------------------------------------------------
__global__ void gat_attn_gemm(const int* __restrict__ adj,
                              const unsigned short* __restrict__ ht,
                              const float* __restrict__ s1, const float* __restrict__ s2,
                              const float* __restrict__ s2maxp,
                              float* __restrict__ out) {
    __shared__ float s2s[NROW];                              // 32 KB
    __shared__ __align__(16) unsigned short bsh[2][NF * 32]; // 2 x 16 KB double buffer

    const int tid  = threadIdx.x;
    const int lane = tid & 31, wave = tid >> 5;
    const int r15 = lane & 15, hi = lane >> 4, g = hi * 8;

    const int rowBase = blockIdx.x * 64 + wave * 16;
    const int row     = rowBase + r15;
    const float s1v   = s1[row];

    // conservative softmax shift: upper bound of the row max (monotone lrelu)
    float km = s1v + *s2maxp;
    km = km > 0.f ? km : ALPHA * km;
    const float maxv = fmaxf(km, 0.f);
    const float e0   = __expf(-maxv);                // coefficient of non-edges
    const int* arow  = adj + row * NROW;

    // prologue: async-stage chunk 0, load s2 into LDS meanwhile
    issue_stage(ht, 0, &bsh[0][0], tid);
    for (int t = tid; t < NROW; t += 128) s2s[t] = s2[t];
    asm volatile("s_wait_asynccnt 0x0" ::: "memory");
    __syncthreads();

    v8f acc[16];
    const v8f z = {0.f, 0.f, 0.f, 0.f, 0.f, 0.f, 0.f, 0.f};
#pragma unroll
    for (int nt = 0; nt < 16; ++nt) acc[nt] = z;
    float densum = 0.f;
    int cur = 0;

    for (int j0 = 0; j0 < NROW; j0 += 32) {
        // issue next chunk into the other buffer while computing this one
        if (j0 + 32 < NROW) issue_stage(ht, j0 + 32, &bsh[cur ^ 1][0], tid);

        // adjacency: 16 values per lane matching the A-fragment K positions
        const int4 A0 = *(const int4*)(arow + j0 + g);
        const int4 A1 = *(const int4*)(arow + j0 + g + 4);
        const int4 A2 = *(const int4*)(arow + j0 + 16 + g);
        const int4 A3 = *(const int4*)(arow + j0 + 16 + g + 4);
        const int adjv[16] = {A0.x, A0.y, A0.z, A0.w, A1.x, A1.y, A1.z, A1.w,
                              A2.x, A2.y, A2.z, A2.w, A3.x, A3.y, A3.z, A3.w};

        // A fragment: elements 0..7 -> K=g+0..7 ; 8..15 -> K=16+g+0..7
        BF16x16 Af;
#pragma unroll
        for (int p = 0; p < 8; ++p) {
            unsigned pk = 0;
#pragma unroll
            for (int hl = 0; hl < 2; ++hl) {
                const int t = 2 * p + hl;
                const int j = j0 + g + (t < 8 ? t : 8 + t);
                float key = s1v + s2s[j];
                key = key > 0.f ? key : ALPHA * key;
                const float c = (adjv[t] > 0) ? __expf(key - maxv) : e0;
                densum += c;                          // softmax denominator, free
                pk |= (unsigned)f2bf(c) << (16 * hl);
            }
            Af.u[p] = pk;
        }

#pragma unroll
        for (int nt = 0; nt < 16; ++nt) {
            BF16x16 Bf;
            const unsigned short* bp = &bsh[cur][0] + (nt * 16 + r15) * 32;
            Bf.q[0] = *(const uint4*)(bp + g);
            Bf.q[1] = *(const uint4*)(bp + 16 + g);
            acc[nt] = __builtin_amdgcn_wmma_f32_16x16x32_bf16(false, Af.v, false, Bf.v,
                                                              (short)0, acc[nt], false, false);
        }

        asm volatile("s_wait_asynccnt 0x0" ::: "memory");  // next chunk landed
        __syncthreads();                                   // all waves done with bsh[cur]
        cur ^= 1;
    }

    // combine the two K-halves of each row's denominator (lanes L and L+16)
    densum += __shfl_xor(densum, 16);

    // route denominators to D-layout rows: acc[nt][r] is row rowBase + 8*hi + r,
    // whose denominator lives in lane (8*hi + r)
    float dinv[8];
#pragma unroll
    for (int r = 0; r < 8; ++r) dinv[r] = 1.0f / __shfl(densum, 8 * hi + r);

#pragma unroll
    for (int nt = 0; nt < 16; ++nt) {
#pragma unroll
        for (int r = 0; r < 8; ++r) {
            float v = acc[nt][r] * dinv[r];
            v = v > 0.f ? v : expm1f(v);
            out[(rowBase + 8 * hi + r) * NF + nt * 16 + r15] = v;
        }
    }
}

// ---------------------------------------------------------------------------
// Launch
// ---------------------------------------------------------------------------
extern "C" void kernel_launch(void* const* d_in, const int* in_sizes, int n_in,
                              void* d_out, int out_size, void* d_ws, size_t ws_size,
                              hipStream_t stream) {
    const int*   adj = (const int*)d_in[0];
    const float* x   = (const float*)d_in[1];
    const float* w   = (const float*)d_in[2];
    const float* a   = (const float*)d_in[3];
    float*       out = (float*)d_out;

    char* ws = (char*)d_ws;
    unsigned short* xb  = (unsigned short*)(ws);                 // 4 MB
    unsigned short* wb  = (unsigned short*)(ws + 4194304);       // 128 KB
    float*          h   = (float*)(ws + 4325376);                // 8 MB
    unsigned short* ht  = (unsigned short*)(ws + 12713984);      // 4 MB
    float*          s1  = (float*)(ws + 16908288);               // 32 KB
    float*          s2  = (float*)(ws + 16941056);               // 32 KB
    float*          s2m = (float*)(ws + 16973824);               // 4 B

    gat_convert  <<<8448, 256, 0, stream>>>(x, w, xb, wb, NROW * NF, NF * NF);
    gat_h_gemm   <<<1024, 256, 0, stream>>>(xb, wb, h, ht);
    gat_scores   <<<1024, 256, 0, stream>>>(h, a, s1, s2);
    gat_s2max    <<<   1, 256, 0, stream>>>(s2, s2m);
    gat_attn_gemm<<< 128, 128, 0, stream>>>(adj, ht, s1, s2, s2m, out);
}
